// MultiHeadAttention_20581483282980
// MI455X (gfx1250) — matrix operand order repaired and checked by reading, not compiled, not run on hardware
//
#include <hip/hip_runtime.h>
#include <hip/hip_bf16.h>
#include <math.h>

typedef __bf16 bf16_t;
typedef __attribute__((ext_vector_type(16))) __bf16 v16bf;
typedef __attribute__((ext_vector_type(8)))  __bf16 v8bf;
typedef __attribute__((ext_vector_type(8)))  float  v8f;
typedef __attribute__((ext_vector_type(4))) unsigned int u32x4;
typedef __attribute__((ext_vector_type(8))) int i32x8;
typedef __attribute__((ext_vector_type(4))) int i32x4;

#define WMMA_BF16(a, b, c) \
  __builtin_amdgcn_wmma_f32_16x16x32_bf16(false, (a), false, (b), (short)0, (c), false, false)

// ---------------------------------------------------------------------------
// Fragment loaders (layouts per CDNA5 ISA §7.12.2)
// A 16x32 bf16: lane<16 -> row=lane, K {k0..k0+7, k0+16..k0+23}
//               lane>=16 -> row=lane-16, K {k0+8..15, k0+24..31}
static __device__ __forceinline__ v16bf load_fragA(const bf16_t* __restrict__ base,
                                                   int ld, int row0, int k0, int lane) {
  const int r    = lane & 15;
  const int half = lane >> 4;
  const bf16_t* p = base + (size_t)(row0 + r) * (size_t)ld + k0 + half * 8;
  v8bf lo = *(const v8bf*)(p);        // K: k0+half*8 .. +7
  v8bf hi = *(const v8bf*)(p + 16);   // K: k0+16+half*8 .. +7
  return __builtin_shufflevector(lo, hi, 0,1,2,3,4,5,6,7,8,9,10,11,12,13,14,15);
}

// B 32x16 bf16 (stored as Bt row-major: row n holds K contiguously):
// lane<16 -> col n=lane, K {k0..k0+7, k0+16..k0+23}; lane>=16 -> n=lane-16, K {k0+8..15, k0+24..31}
static __device__ __forceinline__ v16bf load_fragB(const bf16_t* __restrict__ base,
                                                   int ld, int n0, int k0, int lane) {
  const int n    = lane & 15;
  const int half = lane >> 4;
  const bf16_t* p = base + (size_t)(n0 + n) * (size_t)ld + k0 + half * 8;
  v8bf lo = *(const v8bf*)(p);
  v8bf hi = *(const v8bf*)(p + 16);
  return __builtin_shufflevector(lo, hi, 0,1,2,3,4,5,6,7,8,9,10,11,12,13,14,15);
}

// ---------------------------------------------------------------------------
// Tensor Data Mover: DMA a (128 rows x 32 cols) bf16 tile into LDS.
// D# per CDNA5 ISA. data_size=2B, pad 16B every 64B so
// LDS rows stride 80B (conflict-free ds_load_b128 fragment reads).
static __device__ __forceinline__ uint32_t lds_off(const void* p) {
  return (uint32_t)(uintptr_t)p;   // generic LDS addr: low 32 bits = LDS offset
}

static __device__ __forceinline__ void tdm_load_tile(uint32_t lds_addr,
                                                     const bf16_t* gsrc,
                                                     uint32_t tdim0, uint32_t tdim1,
                                                     uint32_t stride0) {
  const unsigned long long ga = (unsigned long long)(uintptr_t)gsrc;
  u32x4 g0 = { 1u,                                             // count=1, user D#
               lds_addr,                                       // lds_addr[31:0]
               (unsigned)(ga & 0xffffffffull),                 // global_addr[31:0]
               (unsigned)((ga >> 32) & 0x1ffffffull) | 0x80000000u };  // type=2
  // w0: data_size=1(2B)<<16 | pad_enable<<20 | pad_interval=3(64B)<<22 | pad_amount=3(16B)<<25
  i32x8 g1 = { (int)((1u << 16) | (1u << 20) | (3u << 22) | (3u << 25)),
               (int)((tdim0 & 0xffffu) << 16),                 // tensor_dim0[15:0]
               (int)((tdim0 >> 16) | ((tdim1 & 0xffffu) << 16)),
               (int)((tdim1 >> 16) | (32u << 16)),             // tile_dim0 = 32
               (int)128u,                                      // tile_dim1 = 128
               (int)stride0,                                   // tensor_dim0_stride[31:0]
               0, 0 };
  i32x4 z4 = { 0, 0, 0, 0 };
#if defined(__clang_major__) && (__clang_major__ >= 23)
  i32x8 z8 = { 0, 0, 0, 0, 0, 0, 0, 0 };
  __builtin_amdgcn_tensor_load_to_lds(g0, g1, z4, z4, z8, 0);
#else
  __builtin_amdgcn_tensor_load_to_lds(g0, g1, z4, z4, 0);
#endif
}

// ---------------------------------------------------------------------------
__global__ void cvt_f32_bf16(const float* __restrict__ in, bf16_t* __restrict__ out, long n) {
  long i      = (long)blockIdx.x * blockDim.x + threadIdx.x;
  long stride = (long)gridDim.x * blockDim.x;
  for (; i < n; i += stride) out[i] = (bf16_t)in[i];
}

// Vt[(b*16+h)*128 + d][s] = qkv[(b*2048+s)*6144 + 4096 + h*128 + d]
__global__ void transpose_v(const bf16_t* __restrict__ qkv, bf16_t* __restrict__ vt) {
  const int d  = blockIdx.x;        // 0..127
  const int bh = blockIdx.y;        // 0..31
  const int b  = bh >> 4, h = bh & 15;
  bf16_t* dst = vt + ((size_t)bh * 128 + d) * 2048;
  const bf16_t* src = qkv + (size_t)b * 2048 * 6144 + 4096 + h * 128 + d;
  for (int s = threadIdx.x; s < 2048; s += blockDim.x)
    dst[s] = src[(size_t)s * 6144];
}

// ---------------------------------------------------------------------------
// C[M,N] = A[M,K] * Bt[N,K]^T.  4-wave workgroup computes 128x128; A/B tiles
// (128x32 each) are TDM-DMA'd into LDS, double-buffered on TENSORcnt.
template <bool STORE_BF16>
__global__ void __launch_bounds__(128) gemm_wmma(const bf16_t* __restrict__ A,
                                                 const bf16_t* __restrict__ Bt,
                                                 const float* __restrict__ bias,
                                                 bf16_t* __restrict__ Cb,
                                                 float* __restrict__ Cf,
                                                 int M, int N, int K) {
  constexpr int LDT = 40;  // padded LDS row stride in elements (64B data + 16B pad)
  const int lane = threadIdx.x & 31;
  const int wave = threadIdx.x >> 5;        // 0..3
  const int n0   = blockIdx.x * 128;
  const int mblk = blockIdx.y * 128;
  const int m0   = mblk + wave * 32;

  __shared__ __align__(16) bf16_t smem[4][128 * LDT];  // A0 A1 B0 B1 (10 KB each)

  if (wave == 0) {
    tdm_load_tile(lds_off(smem[0]), A  + (size_t)mblk * K, (uint32_t)K, (uint32_t)M, (uint32_t)K);
    tdm_load_tile(lds_off(smem[2]), Bt + (size_t)n0  * K, (uint32_t)K, (uint32_t)N, (uint32_t)K);
  }

  v8f acc[2][8] = {};
  const int nsteps = K / 32;
  for (int s = 0; s < nsteps; ++s) {
    const int cur = s & 1;
    if (wave == 0) __builtin_amdgcn_s_wait_tensorcnt(0);
    __syncthreads();  // tiles [cur] resident; previous buffer fully consumed
    if (wave == 0 && s + 1 < nsteps) {
      const int k1 = (s + 1) * 32;
      tdm_load_tile(lds_off(smem[cur ^ 1]),     A  + (size_t)mblk * K + k1,
                    (uint32_t)K, (uint32_t)M, (uint32_t)K);
      tdm_load_tile(lds_off(smem[2 + (cur ^ 1)]), Bt + (size_t)n0 * K + k1,
                    (uint32_t)K, (uint32_t)N, (uint32_t)K);
    }
    const bf16_t* ab = smem[cur];
    const bf16_t* bb = smem[2 + cur];
    v16bf a0 = load_fragA(ab, LDT, wave * 32,      0, lane);
    v16bf a1 = load_fragA(ab, LDT, wave * 32 + 16, 0, lane);
#pragma unroll
    for (int t = 0; t < 8; ++t) {
      v16bf b = load_fragB(bb, LDT, 16 * t, 0, lane);
      acc[0][t] = WMMA_BF16(a0, b, acc[0][t]);
      acc[1][t] = WMMA_BF16(a1, b, acc[1][t]);
    }
    __syncthreads();  // all waves done with tiles [cur] before step s+2 overwrites
  }

  const int half = lane >> 4;
  const int ln   = lane & 15;
#pragma unroll
  for (int i = 0; i < 2; ++i)
#pragma unroll
    for (int t = 0; t < 8; ++t) {
      const int col = n0 + 16 * t + ln;
      const float bv = STORE_BF16 ? 0.f : bias[col];
#pragma unroll
      for (int r = 0; r < 8; ++r) {
        const int row = m0 + 16 * i + 8 * half + r;
        const float v = acc[i][t][r];
        if (STORE_BF16) Cb[(size_t)row * N + col] = (bf16_t)v;
        else            Cf[(size_t)row * N + col] = v + bv;
      }
    }
}

// ---------------------------------------------------------------------------
// Flash attention: one wave handles (b, h, 16-query tile); key blocks of 32.
__global__ void __launch_bounds__(32) flash_attn(const bf16_t* __restrict__ qkv,
                                                 const bf16_t* __restrict__ vt,
                                                 bf16_t* __restrict__ attn_out) {
  constexpr int S = 2048, E = 6144, HD = 128, DOUT = 2048;
  const float scale = 0.0883883476483184f;  // 1/sqrt(128)
  const int lane = threadIdx.x & 31;
  const int half = lane >> 4, ln = lane & 15;
  const int m0 = blockIdx.x * 16;
  const int h  = blockIdx.y;
  const int b  = blockIdx.z;

  const bf16_t* qbase = qkv + (size_t)b * S * E + h * HD;
  const bf16_t* kbase = qbase + DOUT;  // K block lives at e-offset 2048
  const bf16_t* vtb   = vt + ((size_t)(b * 16 + h)) * HD * S;

  v16bf qf[4];
#pragma unroll
  for (int i = 0; i < 4; ++i) qf[i] = load_fragA(qbase, E, m0, 32 * i, lane);

  v8f acc[8] = {};
  float mrow[8], lrow[8], alpha[8];
#pragma unroll
  for (int r = 0; r < 8; ++r) { mrow[r] = -__builtin_inff(); lrow[r] = 0.f; }

  __shared__ __align__(16) bf16_t pbuf[16 * 32];

  for (int key0 = 0; key0 < m0 + 16; key0 += 32) {
    // S = Q * K^T for 16 queries x 32 keys (K-dim = 128, 4 wmma steps)
    v8f s0 = {}, s1 = {};
#pragma unroll
    for (int i = 0; i < 4; ++i) {
      v16bf kf0 = load_fragB(kbase, E, key0,      32 * i, lane);
      v16bf kf1 = load_fragB(kbase, E, key0 + 16, 32 * i, lane);
      s0 = WMMA_BF16(qf[i], kf0, s0);
      s1 = WMMA_BF16(qf[i], kf1, s1);
    }
    // online softmax: per lane, VGPR slot r corresponds to row m0 + 8*half + r
#pragma unroll
    for (int r = 0; r < 8; ++r) {
      const int row = m0 + 8 * half + r;
      float v0 = (key0 + ln      <= row) ? s0[r] * scale : -__builtin_inff();
      float v1 = (key0 + 16 + ln <= row) ? s1[r] * scale : -__builtin_inff();
      float mx = fmaxf(v0, v1);
#pragma unroll
      for (int off = 1; off < 16; off <<= 1) mx = fmaxf(mx, __shfl_xor(mx, off, 32));
      const float mnew = fmaxf(mrow[r], mx);          // always finite (diag unmasked)
      const float a = __expf(mrow[r] - mnew);
      mrow[r] = mnew;
      const float p0 = __expf(v0 - mnew);
      const float p1 = __expf(v1 - mnew);
      float rs = p0 + p1;
#pragma unroll
      for (int off = 1; off < 16; off <<= 1) rs += __shfl_xor(rs, off, 32);
      lrow[r] = lrow[r] * a + rs;
      alpha[r] = a;
      s0[r] = p0;
      s1[r] = p1;
    }
#pragma unroll
    for (int t = 0; t < 8; ++t)
#pragma unroll
      for (int r = 0; r < 8; ++r) acc[t][r] *= alpha[r];

    // P (C-layout) -> LDS row-major 16x32 bf16, re-read as A-fragment
#pragma unroll
    for (int r = 0; r < 8; ++r) {
      pbuf[(8 * half + r) * 32 + ln]      = (bf16_t)s0[r];
      pbuf[(8 * half + r) * 32 + 16 + ln] = (bf16_t)s1[r];
    }
    __syncthreads();
    v16bf pf = load_fragA(pbuf, 32, 0, 0, lane);
    // O += P (16x32) * V (32x128); Vt row n = head-dim column, contiguous in keys
#pragma unroll
    for (int t = 0; t < 8; ++t) {
      v16bf vf = load_fragB(vtb, S, 16 * t, key0, lane);
      acc[t] = WMMA_BF16(pf, vf, acc[t]);
    }
    __syncthreads();
  }
  // epilogue: divide by l, write [b, s, h*128+d] in bf16 for the out-projection
#pragma unroll
  for (int t = 0; t < 8; ++t)
#pragma unroll
    for (int r = 0; r < 8; ++r) {
      const int row = m0 + 8 * half + r;
      const int col = h * HD + 16 * t + ln;
      attn_out[((size_t)b * S + row) * DOUT + col] = (bf16_t)(acc[t][r] / lrow[r]);
    }
}

// ---------------------------------------------------------------------------
extern "C" void kernel_launch(void* const* d_in, const int* in_sizes, int n_in,
                              void* d_out, int out_size, void* d_ws, size_t ws_size,
                              hipStream_t stream) {
  (void)in_sizes; (void)n_in; (void)out_size; (void)ws_size;
  const float* x     = (const float*)d_in[0];
  const float* w_qkv = (const float*)d_in[1];
  const float* w_out = (const float*)d_in[2];
  const float* b_out = (const float*)d_in[3];
  float* out = (float*)d_out;

  const size_t NX    = (size_t)2 * 2048 * 2048;   //  8388608
  const size_t NWQKV = (size_t)6144 * 2048;       // 12582912
  const size_t NWOUT = (size_t)2048 * 2048;       //  4194304
  const size_t NQKV  = (size_t)2 * 2048 * 6144;   // 25165824
  const size_t NVT   = (size_t)2 * 16 * 128 * 2048;

  bf16_t* xb    = (bf16_t*)d_ws;
  bf16_t* wqkvb = xb + NX;
  bf16_t* woutb = wqkvb + NWQKV;
  bf16_t* qkvb  = woutb + NWOUT;
  bf16_t* vtb   = qkvb + NQKV;
  bf16_t* attnb = vtb + NVT;

  cvt_f32_bf16<<<2048, 256, 0, stream>>>(x,     xb,    (long)NX);
  cvt_f32_bf16<<<2048, 256, 0, stream>>>(w_qkv, wqkvb, (long)NWQKV);
  cvt_f32_bf16<<<2048, 256, 0, stream>>>(w_out, woutb, (long)NWOUT);

  // qkv[b,s,e] = x @ w_qkv^T   (M=4096, N=6144, K=2048)
  gemm_wmma<true><<<dim3(6144 / 128, 4096 / 128), 128, 0, stream>>>(
      xb, wqkvb, nullptr, qkvb, nullptr, 4096, 6144, 2048);

  transpose_v<<<dim3(128, 32), 256, 0, stream>>>(qkvb, vtb);

  flash_attn<<<dim3(2048 / 16, 16, 2), 32, 0, stream>>>(qkvb, vtb, attnb);

  // out = attn @ w_out^T + b_out   (M=4096, N=2048, K=2048)
  gemm_wmma<false><<<dim3(2048 / 128, 4096 / 128), 128, 0, stream>>>(
      attnb, woutb, b_out, nullptr, out, 4096, 2048, 2048);
}
